// Block_36094905155665
// MI455X (gfx1250) — compile-verified, matrix operand order checked by reading, not run on hardware
//
#include <hip/hip_runtime.h>
#include <hip/hip_bf16.h>
#include <math.h>

typedef __attribute__((ext_vector_type(16))) _Float16 v16h;
typedef __attribute__((ext_vector_type(8)))  float    v8f;
typedef __attribute__((ext_vector_type(4)))  unsigned int v4u;
typedef __attribute__((ext_vector_type(8)))  int      v8i;
typedef __attribute__((ext_vector_type(4)))  int      v4i;

#define NB      32          // graphs
#define NPG     1024        // nodes per graph
#define NN      (NB*NPG)    // 32768 nodes
#define DEG     16
#define NE      (NN*DEG)    // 524288 edges
#define ETOT    (NE+NN)     // + self loops
#define INC     128
#define HID     64
#define NH      4
#define FEAT    (NH*HID)    // 256
#define KKEEP   512
#define SLOPE   0.2f

// ISA 7.12.2 16-bit A/B layout: K offset for fragment element j, half-wave lh
__device__ __forceinline__ int koff(int j, int lh) {
    return (j < 8) ? (j + lh * 8) : (j + 8 + lh * 8);
}

// pack two contiguous 8-float runs (f32) into a v16h fragment
__device__ __forceinline__ v16h packA(float4 a, float4 b, float4 c, float4 d) {
    v16h r;
    r[0]  = (_Float16)a.x; r[1]  = (_Float16)a.y; r[2]  = (_Float16)a.z; r[3]  = (_Float16)a.w;
    r[4]  = (_Float16)b.x; r[5]  = (_Float16)b.y; r[6]  = (_Float16)b.z; r[7]  = (_Float16)b.w;
    r[8]  = (_Float16)c.x; r[9]  = (_Float16)c.y; r[10] = (_Float16)c.z; r[11] = (_Float16)c.w;
    r[12] = (_Float16)d.x; r[13] = (_Float16)d.y; r[14] = (_Float16)d.z; r[15] = (_Float16)d.w;
    return r;
}

// ---------------------------------------------------------------- init fills
__global__ void k_init(float* agg, float* m, float* den, int* nmap) {
    int i = blockIdx.x * blockDim.x + threadIdx.x;
    if (i < NN * FEAT) agg[i] = 0.0f;
    if (i < NN * NH) { m[i] = -INFINITY; den[i] = 0.0f; }
    if (i < NN) nmap[i] = -1;
}

// ------------- one-time f32 -> f16 weight convert into FRAGMENT-MAJOR order
// wfsw index: ((kc*16 + colTile)*32 + lane)*16 + j   (kc = K/32 chunk)
// wlsw index: ((kc*4  + colTile)*32 + lane)*16 + j
__global__ void k_cvt(const float* __restrict__ W, const float* __restrict__ W_lin,
                      _Float16* wfsw, _Float16* wlsw) {
    int i = blockIdx.x * blockDim.x + threadIdx.x;
    if (i < INC * FEAT) {
        int j = i & 15, lane = (i >> 4) & 31, t = (i >> 9) & 15, kc = i >> 13;
        int k = kc * 32 + koff(j, lane >> 4);
        wfsw[i] = (_Float16)W[k * FEAT + t * 16 + (lane & 15)];
    }
    if (i < FEAT * HID) {
        int j = i & 15, lane = (i >> 4) & 31, t = (i >> 9) & 3, kc = i >> 11;
        int k = kc * 32 + koff(j, lane >> 4);
        wlsw[i] = (_Float16)W_lin[k * HID + t * 16 + (lane & 15)];
    }
}

// ------------------------------------------------- GEMM1: h = x @ W  (WMMA)
// block = 256 thr (8 waves). Block: 16 rows x 256 cols. wave w -> col tiles 2w,2w+1.
// A tile (16x128 f32) arrives via TDM tensor_load_to_lds; W staged in LDS as
// fragment-major f16 so each B fragment is one aligned 32B LDS read.
__global__ void k_gemm1(const float* __restrict__ x, const _Float16* __restrict__ wf,
                        float* __restrict__ h) {
    __shared__ __align__(64) unsigned char smem[16 * INC * 4 + INC * FEAT * 2];
    float*    ax   = (float*)smem;                         // 16x128 f32, LDS off 0
    _Float16* wlds = (_Float16*)(smem + 16 * INC * 4);     // fragment-major f16

    const int rowBase = blockIdx.x * 16;

    if (threadIdx.x < 32) {   // wave 0 issues the TDM load of the A tile
        unsigned long long ga = (unsigned long long)(const void*)(x + (size_t)rowBase * INC);
        v4u g0 = { 1u,                                   // count=1, user descriptor
                   0u,                                   // lds_addr = 0 (ax)
                   (unsigned)ga,
                   (unsigned)((ga >> 32) & 0x01FFFFFFu) | (2u << 30) };  // type=2
        v8i g1 = { (int)(2u << 16),                      // data_size = 4B
                   (int)(128u << 16),                    // tensor_dim0 = 128
                   (int)0x80000000u,                     // tensor_dim1 = 32768 (lo16<<16)
                   (int)(128u << 16),                    // tile_dim0 = 128
                   16,                                   // tile_dim1 = 16
                   128,                                  // tensor_dim0_stride = 128
                   0, 0 };
        v4i z4 = { 0, 0, 0, 0 };
        v8i z8 = { 0, 0, 0, 0, 0, 0, 0, 0 };
        __builtin_amdgcn_tensor_load_to_lds(g0, g1, z4, z4, z8, 0);
    }
    {   // all waves: bulk-copy swizzled f16 W into LDS (4096 x b128)
        const uint4* src = (const uint4*)wf;
        uint4*       dst = (uint4*)wlds;
        for (int i = threadIdx.x; i < (INC * FEAT * 2) / 16; i += 256) dst[i] = src[i];
    }
    if (threadIdx.x < 32) __builtin_amdgcn_s_wait_tensorcnt(0);
    __syncthreads();

    const int wave = threadIdx.x >> 5;
    const int lane = threadIdx.x & 31;
    const int lh = lane >> 4, lm = lane & 15;
    const int t0 = wave * 2, t1 = wave * 2 + 1;
    const int col0 = t0 * 16 + lm;
    const int col1 = col0 + 16;
    v8f acc0 = {}; v8f acc1 = {};
    for (int kk = 0; kk < INC; kk += 32) {
        const float4* r0 = (const float4*)&ax[lm * INC + kk + lh * 8];       // K: lh*8..+7
        const float4* r1 = (const float4*)&ax[lm * INC + kk + 16 + lh * 8];  // K: 16+lh*8..+7
        v16h af = packA(r0[0], r0[1], r1[0], r1[1]);
        int kc = kk >> 5;
        v16h b0 = *(const v16h*)&wlds[(((kc * 16) + t0) * 32 + lane) * 16];
        v16h b1 = *(const v16h*)&wlds[(((kc * 16) + t1) * 32 + lane) * 16];
        acc0 = __builtin_amdgcn_wmma_f32_16x16x32_f16(false, af, false, b0,
                                                      (short)0, acc0, false, false);
        acc1 = __builtin_amdgcn_wmma_f32_16x16x32_f16(false, af, false, b1,
                                                      (short)0, acc1, false, false);
    }
    for (int r = 0; r < 8; ++r) {                      // C/D: VGPR r -> M=r / r+8
        int row = rowBase + r + lh * 8;
        h[row * FEAT + col0] = acc0[r];
        h[row * FEAT + col1] = acc1[r];
    }
}

// ----------------- per-node attention logits: one wave per node, shfl reduce
__global__ void k_al(const float* __restrict__ h, const float* __restrict__ a_src,
                     const float* __restrict__ a_dst, float* als, float* ald) {
    int gid = blockIdx.x * blockDim.x + threadIdx.x;
    int n = gid >> 5, lane = gid & 31;
    if (n >= NN) return;
    const float* hv = h + (size_t)n * FEAT;
    for (int hh = 0; hh < NH; ++hh) {
        float v0 = hv[hh * HID + lane], v1 = hv[hh * HID + 32 + lane];
        float s = v0 * a_src[hh * HID + lane] + v1 * a_src[hh * HID + 32 + lane];
        float d = v0 * a_dst[hh * HID + lane] + v1 * a_dst[hh * HID + 32 + lane];
        for (int off = 16; off > 0; off >>= 1) {
            s += __shfl_down(s, off);
            d += __shfl_down(d, off);
        }
        if (lane == 0) { als[n * NH + hh] = s; ald[n * NH + hh] = d; }
    }
}

// ---------------------------------- edge pass 1: e = leakyrelu, segment max
__global__ void k_att1(const int* __restrict__ ei, const float* __restrict__ als,
                       const float* __restrict__ ald, float* ebuf, float* m) {
    int idx = blockIdx.x * blockDim.x + threadIdx.x;
    if (idx >= ETOT) return;
    int s, d;
    if (idx < NE) { s = ei[idx]; d = ei[NE + idx]; } else { s = d = idx - NE; }
    for (int hh = 0; hh < NH; ++hh) {
        float v = als[s * NH + hh] + ald[d * NH + hh];
        v = v > 0.f ? v : SLOPE * v;
        ebuf[idx * NH + hh] = v;
        atomicMax(&m[d * NH + hh], v);
    }
}

// ---------------------------------- edge pass 2: exp, segment sum
__global__ void k_att2(const int* __restrict__ ei, const float* __restrict__ m,
                       float* ebuf, float* den) {
    int idx = blockIdx.x * blockDim.x + threadIdx.x;
    if (idx >= ETOT) return;
    int d = (idx < NE) ? ei[NE + idx] : idx - NE;
    for (int hh = 0; hh < NH; ++hh) {
        float ex = expf(ebuf[idx * NH + hh] - m[d * NH + hh]);
        ebuf[idx * NH + hh] = ex;
        atomicAdd(&den[d * NH + hh], ex);
    }
}

// ------------- edge pass 3: weighted scatter-add, one wave per edge (coalesced)
__global__ void k_att3(const int* __restrict__ ei, const float* __restrict__ h,
                       const float* __restrict__ ebuf, const float* __restrict__ den,
                       float* agg) {
    int gid = blockIdx.x * blockDim.x + threadIdx.x;
    int idx = gid >> 5, lane = gid & 31;
    if (idx >= ETOT) return;
    int s, d;
    if (idx < NE) { s = ei[idx]; d = ei[NE + idx]; } else { s = d = idx - NE; }
    float a0 = ebuf[idx * NH + 0] / den[d * NH + 0];
    float a1 = ebuf[idx * NH + 1] / den[d * NH + 1];
    float a2 = ebuf[idx * NH + 2] / den[d * NH + 2];
    float a3 = ebuf[idx * NH + 3] / den[d * NH + 3];
    const float* hv = h + (size_t)s * FEAT;
    float* ov = agg + (size_t)d * FEAT;
#pragma unroll
    for (int i = 0; i < 8; ++i) {           // c = lane + 32*i; head = i>>1
        int c = lane + 32 * i;
        float av = (i < 2) ? a0 : (i < 4) ? a1 : (i < 6) ? a2 : a3;
        atomicAdd(&ov[c], hv[c] * av);
    }
}

// --------------------------- GEMM2: xl = (agg + b_gat) @ W_lin + b_lin (WMMA)
// block = 256 thr (8 waves). Block: 32 rows x 64 cols; W_lin fragment-major in LDS.
__global__ void k_gemm2(const float* __restrict__ agg, const float* __restrict__ b_gat,
                        const _Float16* __restrict__ wl, const float* __restrict__ b_lin,
                        float* __restrict__ xl) {
    __shared__ __align__(64) unsigned char smem[32 * FEAT * 4 + FEAT * HID * 2];
    float*    ax = (float*)smem;                      // 32x256 f32
    _Float16* wb = (_Float16*)(smem + 32 * FEAT * 4); // fragment-major f16
    const int rowBase = blockIdx.x * 32;
    for (int i = threadIdx.x; i < 32 * FEAT; i += 256) {
        int r = i >> 8, c = i & 255;
        ax[i] = agg[(size_t)(rowBase + r) * FEAT + c] + b_gat[c];
    }
    {
        const uint4* src = (const uint4*)wl;
        uint4*       dst = (uint4*)wb;
        for (int i = threadIdx.x; i < (FEAT * HID * 2) / 16; i += 256) dst[i] = src[i];
    }
    __syncthreads();
    const int wave = threadIdx.x >> 5;
    const int lane = threadIdx.x & 31;
    const int lh = lane >> 4, lm = lane & 15;
    const int rOff = (wave >> 2) * 16;
    const int tc = wave & 3;
    v8f acc = {};
    for (int kk = 0; kk < FEAT; kk += 32) {
        const float4* r0 = (const float4*)&ax[(rOff + lm) * FEAT + kk + lh * 8];
        const float4* r1 = (const float4*)&ax[(rOff + lm) * FEAT + kk + 16 + lh * 8];
        v16h af = packA(r0[0], r0[1], r1[0], r1[1]);
        int kc = kk >> 5;
        v16h bf = *(const v16h*)&wb[((kc * 4 + tc) * 32 + lane) * 16];
        acc = __builtin_amdgcn_wmma_f32_16x16x32_f16(false, af, false, bf,
                                                     (short)0, acc, false, false);
    }
    for (int r = 0; r < 8; ++r) {
        int row = rowBase + rOff + r + lh * 8;
        xl[row * HID + tc * 16 + lm] = acc[r] + b_lin[tc * 16 + lm];
    }
}

// ---------------------------- score: one wave per node, shfl reduce
__global__ void k_score(const float* __restrict__ xl, const float* __restrict__ p,
                        float* sc) {
    int gid = blockIdx.x * blockDim.x + threadIdx.x;
    int n = gid >> 5, lane = gid & 31;
    if (n >= NN) return;
    float p0 = p[lane], p1 = p[32 + lane];
    float nrm = p0 * p0 + p1 * p1;
    float s = xl[(size_t)n * HID + lane] * p0 + xl[(size_t)n * HID + 32 + lane] * p1;
    for (int off = 16; off > 0; off >>= 1) {
        s   += __shfl_down(s, off);
        nrm += __shfl_down(nrm, off);
    }
    if (lane == 0) sc[n] = tanhf(s / sqrtf(nrm));
}

// ------------------------------- top-K per graph: bitonic sort 1024 in LDS
__global__ void k_topk(const float* __restrict__ sc, float* topv, int* perm,
                       int* nmap, float* batch_out) {
    __shared__ float sv[NPG];
    __shared__ int   si[NPG];
    const int b = blockIdx.x, t = threadIdx.x;
    sv[t] = sc[b * NPG + t]; si[t] = t;
    __syncthreads();
    for (int ks = 2; ks <= NPG; ks <<= 1) {
        for (int j = ks >> 1; j > 0; j >>= 1) {
            int ixj = t ^ j;
            if (ixj > t) {
                bool up = (t & ks) == 0;   // up segments hold descending order
                float v1 = sv[t], v2 = sv[ixj];
                int i1 = si[t], i2 = si[ixj];
                bool pre = (v1 > v2) || (v1 == v2 && i1 < i2);  // t should precede
                if (up ? !pre : pre) { sv[t] = v2; sv[ixj] = v1; si[t] = i2; si[ixj] = i1; }
            }
            __syncthreads();
        }
    }
    if (t < KKEEP) {
        int g = b * NPG + si[t];
        int o = b * KKEEP + t;
        perm[o] = g;
        topv[o] = sv[t];
        nmap[g] = o;
        batch_out[o] = (float)b;
    }
}

// -------------------------- gather+gate + per-graph mean/max pool
__global__ void k_pool(const float* __restrict__ xl, const int* __restrict__ perm,
                       const float* __restrict__ topv, float* xp, float* x1) {
    const int b = blockIdx.x, c = threadIdx.x;   // 64 threads
    float sum = 0.f, mx = -INFINITY;
    for (int k = 0; k < KKEEP; ++k) {
        int o = b * KKEEP + k;
        float v = xl[(size_t)perm[o] * HID + c] * topv[o];
        xp[(size_t)o * HID + c] = v;
        sum += v; mx = fmaxf(mx, v);
    }
    x1[b * (2 * HID) + c] = sum / (float)KKEEP;
    x1[b * (2 * HID) + HID + c] = mx;
}

// -------------------------------------------------- edge remap / filter
__global__ void k_remap(const int* __restrict__ ei, const float* __restrict__ edge,
                        const int* __restrict__ nmap, float* eidx_out, float* edge_out) {
    int e = blockIdx.x * blockDim.x + threadIdx.x;
    if (e >= NE) return;
    int ns = nmap[ei[e]], nd = nmap[ei[NE + e]];
    bool valid = (ns >= 0) && (nd >= 0);
    eidx_out[e]      = (float)(valid ? ns : -1);
    eidx_out[NE + e] = (float)(valid ? nd : -1);
    const float4* es = (const float4*)(edge + (size_t)e * 8);
    float4*       ed = (float4*)(edge_out + (size_t)e * 8);
    float4 zz = make_float4(0.f, 0.f, 0.f, 0.f);
    ed[0] = valid ? es[0] : zz;
    ed[1] = valid ? es[1] : zz;
}

extern "C" void kernel_launch(void* const* d_in, const int* in_sizes, int n_in,
                              void* d_out, int out_size, void* d_ws, size_t ws_size,
                              hipStream_t stream) {
    const float* x     = (const float*)d_in[0];
    const int*   ei    = (const int*)  d_in[1];   // [2, E]
    const float* edge  = (const float*)d_in[2];
    const float* W     = (const float*)d_in[4];   // [128, 256] flat
    const float* a_src = (const float*)d_in[5];
    const float* a_dst = (const float*)d_in[6];
    const float* b_gat = (const float*)d_in[7];
    const float* W_lin = (const float*)d_in[8];
    const float* b_lin = (const float*)d_in[9];
    const float* p     = (const float*)d_in[10];

    float* ws = (float*)d_ws;
    size_t o = 0;
    float* h     = ws + o; o += (size_t)NN * FEAT;
    float* agg   = ws + o; o += (size_t)NN * FEAT;
    float* als   = ws + o; o += (size_t)NN * NH;
    float* ald   = ws + o; o += (size_t)NN * NH;
    float* ebuf  = ws + o; o += (size_t)ETOT * NH;
    float* m     = ws + o; o += (size_t)NN * NH;
    float* den   = ws + o; o += (size_t)NN * NH;
    float* xl    = ws + o; o += (size_t)NN * HID;
    float* sc    = ws + o; o += (size_t)NN;
    float* topv  = ws + o; o += (size_t)NB * KKEEP;
    int*   perm  = (int*)(ws + o); o += (size_t)NB * KKEEP;
    int*   nmap  = (int*)(ws + o); o += (size_t)NN;
    _Float16* wf16   = (_Float16*)(ws + o); o += (size_t)(INC * FEAT) / 2;
    _Float16* wlin16 = (_Float16*)(ws + o); o += (size_t)(FEAT * HID) / 2;

    float* out = (float*)d_out;
    float* o_xp    = out;                                    // B*K*HID
    float* o_eidx  = o_xp   + (size_t)NB * KKEEP * HID;      // 2*E
    float* o_edge  = o_eidx + (size_t)2 * NE;                // E*8
    float* o_batch = o_edge + (size_t)NE * 8;                // B*K
    float* o_x1    = o_batch + (size_t)NB * KKEEP;           // B*2*HID

    k_init <<<(NN * FEAT + 255) / 256, 256, 0, stream>>>(agg, m, den, nmap);
    k_cvt  <<<(INC * FEAT + 255) / 256, 256, 0, stream>>>(W, W_lin, wf16, wlin16);
    k_gemm1<<<NN / 16, 256, 0, stream>>>(x, wf16, h);
    k_al   <<<(NN * 32 + 255) / 256, 256, 0, stream>>>(h, a_src, a_dst, als, ald);
    k_att1 <<<(ETOT + 255) / 256, 256, 0, stream>>>(ei, als, ald, ebuf, m);
    k_att2 <<<(ETOT + 255) / 256, 256, 0, stream>>>(ei, m, ebuf, den);
    k_att3 <<<(ETOT * 32 + 255) / 256, 256, 0, stream>>>(ei, h, ebuf, den, agg);
    k_gemm2<<<NN / 32, 256, 0, stream>>>(agg, b_gat, wlin16, b_lin, xl);
    k_score<<<(NN * 32 + 255) / 256, 256, 0, stream>>>(xl, p, sc);
    k_topk <<<NB, NPG, 0, stream>>>(sc, topv, perm, nmap, o_batch);
    k_pool <<<NB, HID, 0, stream>>>(xl, perm, topv, o_xp, o_x1);
    k_remap<<<(NE + 255) / 256, 256, 0, stream>>>(ei, edge, nmap, o_eidx, o_edge);
}
